// DeepSeekMoELayer_57054345560341
// MI455X (gfx1250) — matrix operand order repaired and checked by reading, not compiled, not run on hardware
//
#include <hip/hip_runtime.h>
#include <hip/hip_bf16.h>
#include <math.h>

typedef __attribute__((ext_vector_type(16))) _Float16 v16h;
typedef __attribute__((ext_vector_type(8)))  _Float16 v8h;
typedef __attribute__((ext_vector_type(8)))  float    v8f;
typedef __attribute__((ext_vector_type(4)))  int      v4i;

#define HIDDEN 1024
#define TOKENS 16384
#define NUM_SHARED 2
#define NUM_ROUTED 6
#define NUM_TOTAL  8

#if defined(__AMDGCN__) && __has_builtin(__builtin_amdgcn_global_load_async_to_lds_b128)
#define HAVE_ASYNC_LDS 1
#else
#define HAVE_ASYNC_LDS 0
#endif

#define AS1 __attribute__((address_space(1)))
#define AS3 __attribute__((address_space(3)))

#if HAVE_ASYNC_LDS
__device__ __forceinline__ void async_copy_b128(const _Float16* g, _Float16* l) {
    __builtin_amdgcn_global_load_async_to_lds_b128(
        (AS1 v4i*)g, (AS3 v4i*)l, 0, 0);
}
__device__ __forceinline__ void wait_async_zero() {
#if __has_builtin(__builtin_amdgcn_s_wait_asynccnt)
    __builtin_amdgcn_s_wait_asynccnt(0);
#else
    asm volatile("s_wait_asynccnt 0x0" ::: "memory");
#endif
}
#endif

// ---------------------------------------------------------------------------
// fp32 -> f16 conversion (grid-stride), for activations x
// ---------------------------------------------------------------------------
__global__ void cvt_f32_to_f16_kernel(const float* __restrict__ src,
                                      _Float16* __restrict__ dst, int n) {
    int i = blockIdx.x * blockDim.x + threadIdx.x;
    int stride = gridDim.x * blockDim.x;
    for (; i < n; i += stride) dst[i] = (_Float16)src[i];
}

// ---------------------------------------------------------------------------
// Tiled transpose + convert: dst[e][n][k] = (f16) src[e][k][n]   (H x H)
// ---------------------------------------------------------------------------
__global__ void transpose_cvt_kernel(const float* __restrict__ src,
                                     _Float16* __restrict__ dst) {
    constexpr int H = HIDDEN;
    __shared__ float tile[32][33];
    const size_t e = blockIdx.z;
    const float* s = src + e * H * H;
    _Float16* d = dst + e * H * H;
    const int k0 = blockIdx.y * 32;
    const int n0 = blockIdx.x * 32;
    const int tx = threadIdx.x;
#pragma unroll
    for (int i = threadIdx.y; i < 32; i += 8)
        tile[i][tx] = s[(size_t)(k0 + i) * H + n0 + tx];
    __syncthreads();
#pragma unroll
    for (int i = threadIdx.y; i < 32; i += 8)
        d[(size_t)(n0 + i) * H + k0 + tx] = (_Float16)tile[tx][i];
}

// ---------------------------------------------------------------------------
// Router: one wave per token; softmax over 6 logits, top-2 mask -> wts[b][e]
// ---------------------------------------------------------------------------
__global__ void router_kernel(const float* __restrict__ x,
                              const float* __restrict__ rw,
                              const float* __restrict__ rb,
                              float* __restrict__ wts) {
    const int lane = threadIdx.x & 31;
    const int wave = threadIdx.x >> 5;
    const int b = blockIdx.x * 8 + wave;
    if (b >= TOKENS) return;

    float p[NUM_ROUTED];
#pragma unroll
    for (int e = 0; e < NUM_ROUTED; e++) p[e] = 0.f;

    const float* xr = x + (size_t)b * HIDDEN;
    for (int h = lane; h < HIDDEN; h += 32) {
        float xv = xr[h];
        const float* r = rw + h * NUM_ROUTED;
#pragma unroll
        for (int e = 0; e < NUM_ROUTED; e++) p[e] += xv * r[e];
    }
#pragma unroll
    for (int off = 16; off > 0; off >>= 1) {
#pragma unroll
        for (int e = 0; e < NUM_ROUTED; e++)
            p[e] += __shfl_down(p[e], off, 32);
    }
    if (lane == 0) {
#pragma unroll
        for (int e = 0; e < NUM_ROUTED; e++) p[e] += rb[e];
        float mx = p[0];
#pragma unroll
        for (int e = 1; e < NUM_ROUTED; e++) mx = fmaxf(mx, p[e]);
        float s = 0.f;
#pragma unroll
        for (int e = 0; e < NUM_ROUTED; e++) { p[e] = expf(p[e] - mx); s += p[e]; }
        float inv = 1.f / s;
#pragma unroll
        for (int e = 0; e < NUM_ROUTED; e++) p[e] *= inv;
        int i1 = 0;
#pragma unroll
        for (int e = 1; e < NUM_ROUTED; e++) if (p[e] > p[i1]) i1 = e;
        int i2 = (i1 == 0) ? 1 : 0;
#pragma unroll
        for (int e = 0; e < NUM_ROUTED; e++)
            if (e != i1 && p[e] > p[i2]) i2 = e;
        float* wo = wts + (size_t)b * NUM_ROUTED;
#pragma unroll
        for (int e = 0; e < NUM_ROUTED; e++)
            wo[e] = (e == i1 || e == i2) ? p[e] : 0.f;
    }
}

// Branch-free tanh via hw transcendentals: 1 - 2/(exp2(2*log2e*x)+1)
__device__ __forceinline__ float fast_tanh(float x) {
    float t = __builtin_amdgcn_exp2f(x * 2.8853900817779268f);
    return 1.0f - 2.0f * __builtin_amdgcn_rcpf(t + 1.0f);
}

// ---------------------------------------------------------------------------
// WMMA GEMM: C[M,N] = A_f16[M,K] @ WT_f16[N,K]^T   (M=16384, N=K=1024)
// Block tile 128x128, BK=64, 256 threads = 8 waves (2x4), wave tile 64x32.
// LDS double-buffered; staging via GLOBAL_LOAD_ASYNC_TO_LDS_B128 when
// available (ASYNCcnt-tracked), else register-staged b128 path.
// MODE 0: outH = (f16) tanh(acc + bias)          (layer 1 -> hmid)
// MODE 1: outF (+)= tokw * (acc + bias)  (fp32)  (layer 2 -> final out)
// ---------------------------------------------------------------------------
template <int MODE>
__global__ __launch_bounds__(256) void moe_gemm_kernel(
    const _Float16* __restrict__ A, const _Float16* __restrict__ WT,
    const float* __restrict__ bias,
    _Float16* __restrict__ outH, float* __restrict__ outF,
    const float* __restrict__ tokw, int expert, int init) {

    constexpr int H = HIDDEN;
    constexpr int BM = 128, BN = 128, BK = 64;
    constexpr int LD = 72;            // padded row stride (halves)
    constexpr int NSTAGE = H / BK;    // 16

    __shared__ __align__(16) _Float16 As[2][BM * LD];  // [row][k]
    __shared__ __align__(16) _Float16 Bs[2][BN * LD];  // [col][k]

    const int tid  = threadIdx.x;
    const int lane = tid & 31;
    const int wave = tid >> 5;
    const int wm = wave >> 2;      // 0..1
    const int wn = wave & 3;       // 0..3
    const int l16 = lane & 15;
    const int lhalf = lane >> 4;   // 0/1

    const int bm = blockIdx.y * BM;
    const int bn = blockIdx.x * BN;

    v8f acc[4][2];
#pragma unroll
    for (int mt = 0; mt < 4; mt++)
#pragma unroll
        for (int nt = 0; nt < 2; nt++)
#pragma unroll
            for (int i = 0; i < 8; i++) acc[mt][nt][i] = 0.f;

    const _Float16* Ab  = A  + (size_t)bm * H;
    const _Float16* WTb = WT + (size_t)bn * H;

    // staging map: 1024 chunks of 8 halves per tile, 4 chunks per thread
    const int srow = tid >> 1;            // rows 0..127 (2 threads per row)
    const int skc0 = (tid & 1) << 5;      // half offsets 0 / 32

    auto compute = [&](int buf) {
#pragma unroll
        for (int ks = 0; ks < BK; ks += 32) {
            // A frag: lanes 0-15: K 0-7 (V0-3), K 16-23 (V4-7); lanes 16-31: +8
            v16h af[4];
#pragma unroll
            for (int mt = 0; mt < 4; mt++) {
                const _Float16* p =
                    &As[buf][(wm * 64 + mt * 16 + l16) * LD + ks + lhalf * 8];
                v8h lo = *(const v8h*)p;
                v8h hi = *(const v8h*)(p + 16);
#pragma unroll
                for (int j = 0; j < 8; j++) { af[mt][j] = lo[j]; af[mt][j + 8] = hi[j]; }
            }
            // B frag: lanes 0-15: K 0-7, K 16-23 of col l16; lanes 16-31: +8
            v16h bf[2];
#pragma unroll
            for (int nt = 0; nt < 2; nt++) {
                const _Float16* p =
                    &Bs[buf][(wn * 32 + nt * 16 + l16) * LD + ks + lhalf * 8];
                v8h lo = *(const v8h*)p;
                v8h hi = *(const v8h*)(p + 16);
#pragma unroll
                for (int j = 0; j < 8; j++) { bf[nt][j] = lo[j]; bf[nt][j + 8] = hi[j]; }
            }
#pragma unroll
            for (int mt = 0; mt < 4; mt++)
#pragma unroll
                for (int nt = 0; nt < 2; nt++)
                    acc[mt][nt] = __builtin_amdgcn_wmma_f32_16x16x32_f16(
                        false, af[mt], false, bf[nt], (short)0, acc[mt][nt],
                        false, false);
        }
    };

#if HAVE_ASYNC_LDS
    auto stage = [&](int kt, int buf) {
#pragma unroll
        for (int i = 0; i < 4; i++) {
            int kc = skc0 + i * 8;
            async_copy_b128(Ab  + (size_t)srow * H + kt + kc,
                            &As[buf][srow * LD + kc]);
            async_copy_b128(WTb + (size_t)srow * H + kt + kc,
                            &Bs[buf][srow * LD + kc]);
        }
    };
    stage(0, 0);
    for (int s = 0; s < NSTAGE; s++) {
        wait_async_zero();          // this wave's async writes done
        __syncthreads();            // everyone's writes visible
        const int buf = s & 1;
        if (s + 1 < NSTAGE) stage((s + 1) * BK, buf ^ 1);
        compute(buf);
    }
#else
    v8h ra[4], rb[4];
    auto load_global = [&](int kt) {
#pragma unroll
        for (int i = 0; i < 4; i++) {
            int kc = skc0 + i * 8;
            ra[i] = *(const v8h*)(Ab  + (size_t)srow * H + kt + kc);
            rb[i] = *(const v8h*)(WTb + (size_t)srow * H + kt + kc);
        }
    };
    auto store_lds = [&](int buf) {
#pragma unroll
        for (int i = 0; i < 4; i++) {
            int kc = skc0 + i * 8;
            *(v8h*)(&As[buf][srow * LD + kc]) = ra[i];
            *(v8h*)(&Bs[buf][srow * LD + kc]) = rb[i];
        }
    };
    load_global(0);
    store_lds(0);
    for (int s = 0; s < NSTAGE; s++) {
        __syncthreads();
        const int buf = s & 1;
        const bool pf = (s + 1) < NSTAGE;
        if (pf) load_global((s + 1) * BK);
        compute(buf);
        if (pf) store_lds(buf ^ 1);
    }
#endif

    // ---- epilogue. C layout: VGPR i -> M = i + 8*lhalf, N = l16
#pragma unroll
    for (int mt = 0; mt < 4; mt++) {
#pragma unroll
        for (int nt = 0; nt < 2; nt++) {
            const int col = bn + wn * 32 + nt * 16 + l16;
            const float bcol = bias[col];
#pragma unroll
            for (int i = 0; i < 8; i++) {
                const int row = bm + wm * 64 + mt * 16 + lhalf * 8 + i;
                float v = acc[mt][nt][i] + bcol;
                if (MODE == 0) {
                    outH[(size_t)row * H + col] = (_Float16)fast_tanh(v);
                } else {
                    float s = tokw ? tokw[(size_t)row * NUM_ROUTED + expert] : 1.0f;
                    v *= s;
                    size_t idx = (size_t)row * H + col;
                    if (!init) v += outF[idx];
                    outF[idx] = v;
                }
            }
        }
    }
}

// ---------------------------------------------------------------------------
extern "C" void kernel_launch(void* const* d_in, const int* in_sizes, int n_in,
                              void* d_out, int out_size, void* d_ws, size_t ws_size,
                              hipStream_t stream) {
    const float* x          = (const float*)d_in[0];
    const float* shared_w1  = (const float*)d_in[1];
    const float* shared_b1  = (const float*)d_in[2];
    const float* shared_w2  = (const float*)d_in[3];
    const float* shared_b2  = (const float*)d_in[4];
    const float* routed_w1  = (const float*)d_in[5];
    const float* routed_b1  = (const float*)d_in[6];
    const float* routed_w2  = (const float*)d_in[7];
    const float* routed_b2  = (const float*)d_in[8];
    const float* router_w   = (const float*)d_in[9];
    const float* router_b   = (const float*)d_in[10];
    float* out = (float*)d_out;

    const size_t H = HIDDEN, B = TOKENS;
    char* ws = (char*)d_ws;
    _Float16* x16    = (_Float16*)(ws);                             // 32 MiB
    _Float16* wt1_16 = (_Float16*)(ws + 33554432);                  // 16 MiB (8 experts, transposed)
    _Float16* wt2_16 = (_Float16*)(ws + 33554432 + 16777216);       // 16 MiB
    _Float16* hmid   = (_Float16*)(ws + 33554432 + 2 * 16777216);   // 32 MiB
    float*    wts    = (float*)(ws + 2 * 33554432 + 2 * 16777216);  // 384 KiB

    {
        int n = (int)(B * H);
        int blocks = (n + 2047) / 2048;
        if (blocks > 8192) blocks = 8192;
        cvt_f32_to_f16_kernel<<<blocks, 256, 0, stream>>>(x, x16, n);
    }
    dim3 tb(32, 8);
    transpose_cvt_kernel<<<dim3(32, 32, NUM_SHARED), tb, 0, stream>>>(shared_w1, wt1_16);
    transpose_cvt_kernel<<<dim3(32, 32, NUM_ROUTED), tb, 0, stream>>>(routed_w1, wt1_16 + NUM_SHARED * H * H);
    transpose_cvt_kernel<<<dim3(32, 32, NUM_SHARED), tb, 0, stream>>>(shared_w2, wt2_16);
    transpose_cvt_kernel<<<dim3(32, 32, NUM_ROUTED), tb, 0, stream>>>(routed_w2, wt2_16 + NUM_SHARED * H * H);

    router_kernel<<<TOKENS / 8, 256, 0, stream>>>(x, router_w, router_b, wts);

    dim3 ggrid(HIDDEN / 128, TOKENS / 128);  // (N tiles, M tiles)
    for (int e = 0; e < NUM_TOTAL; e++) {
        const _Float16* w1 = wt1_16 + (size_t)e * H * H;
        const _Float16* w2 = wt2_16 + (size_t)e * H * H;
        const float* b1 = (e < NUM_SHARED) ? shared_b1 + (size_t)e * H
                                           : routed_b1 + (size_t)(e - NUM_SHARED) * H;
        const float* b2 = (e < NUM_SHARED) ? shared_b2 + (size_t)e * H
                                           : routed_b2 + (size_t)(e - NUM_SHARED) * H;
        const float* wptr = (e < NUM_SHARED) ? nullptr : wts;
        int eidx = (e < NUM_SHARED) ? 0 : (e - NUM_SHARED);
        int init = (e == 0) ? 1 : 0;

        moe_gemm_kernel<0><<<ggrid, 256, 0, stream>>>(
            x16, w1, b1, hmid, nullptr, nullptr, 0, 0);
        moe_gemm_kernel<1><<<ggrid, 256, 0, stream>>>(
            hmid, w2, b2, nullptr, out, wptr, eidx, init);
    }
}
